// GroupedTransformerEncoderLayer_77644418777373
// MI455X (gfx1250) — matrix operand order repaired and checked by reading, not compiled, not run on hardware
//
#include <hip/hip_runtime.h>
#include <hip/hip_bf16.h>
#include <math.h>

// ---------------------------------------------------------------------------
// Types for CDNA5 WMMA (wave32, 16x16x32 bf16 -> f32)
// ---------------------------------------------------------------------------
typedef __bf16 bf16_t;
typedef bf16_t v16bf __attribute__((ext_vector_type(16)));
typedef float  v8f   __attribute__((ext_vector_type(8)));
typedef float  f32x4 __attribute__((ext_vector_type(4)));
typedef unsigned int u32x4 __attribute__((ext_vector_type(4)));
typedef unsigned int u32x2 __attribute__((ext_vector_type(2)));

// pack two fp32 -> two bf16 (RNE) in one u32: [31:16]=y, [15:0]=x
__device__ inline unsigned pk2bf(float x, float y) {
    union { float f; unsigned u; } a, b; a.f = x; b.f = y;
    unsigned lo = (a.u + 0x7FFFu + ((a.u >> 16) & 1u)) >> 16;
    unsigned hi = (b.u + 0x7FFFu + ((b.u >> 16) & 1u)) & 0xFFFF0000u;
    return hi | lo;
}

#define BM       64
#define BN       128
#define KSTEP    32
#define LSTRIDE  40   // 32 + 8 pad; 80B rows keep 16B alignment and spread banks

// ---------------------------------------------------------------------------
// Tiled GEMM: C[b] = alpha * A[b] @ B[b](^T) + bias, optional ReLU.
// Block = 128 threads (4 waves). Block tile 64x128; each wave owns a 32x64
// tile = 2x4 WMMA fragments (8 v_wmma per K-step, A frags reused across 4 B).
// All dims are multiples of BM/BN/KSTEP here: no masking.
// ---------------------------------------------------------------------------
template<bool TRANSB, bool RELU>
__global__ __launch_bounds__(128)
void gemm_wmma_bf16(const float* __restrict__ A, long lda, long strideA,
                    const float* __restrict__ B, long ldb, long strideB,
                    const float* __restrict__ bias,
                    float* __restrict__ C, long ldc, long strideC,
                    int M, int Ncols, int Kdim, float alpha)
{
    __shared__ bf16_t As[BM][LSTRIDE];   // As[m][k]
    __shared__ bf16_t Bs[BN][LSTRIDE];   // Bs[n][k]  (K-contiguous per output col)

    const int t    = threadIdx.x;
    const int lane = t & 31;
    const int wave = t >> 5;
    const int wr   = wave >> 1;            // wave row (0..1): 32-row slice
    const int wc   = wave & 1;             // wave col (0..1): 64-col slice
    const int rsel = lane >> 4;
    const int rlo  = lane & 15;

    const int batch = blockIdx.z;
    const float* Ab = A + (long)batch * strideA;
    const float* Bb = B + (long)batch * strideB;
    float*       Cb = C + (long)batch * strideC;

    const int m0 = blockIdx.y * BM;
    const int n0 = blockIdx.x * BN;

    // ---- running global pointers: all muls hoisted out of the K loop ----
    // A: 64 rows x 32 K-cols per step; thread handles row (t>>3), float4 col (t&7)
    const float* aRun = Ab + (long)(m0 + (t >> 3)) * lda + ((t & 7) * 4);
    const long   lda16 = 16 * lda;
    // B staging pointers
    const float* bRun;
    long bStepPass, bStepK;
    if (TRANSB) {  // B is (Ncols x K) row-major: rows like A
        bRun      = Bb + (long)(n0 + (t >> 3)) * ldb + ((t & 7) * 4);
        bStepPass = 16 * ldb;   // +16 rows per pass (8 passes -> 128 rows)
        bStepK    = KSTEP;      // advance K
    } else {       // B is (K x Ncols): thread K-pair 2*(t>>5), float4 col (t&31)
        bRun      = Bb + (long)((t >> 5) * 2) * ldb + n0 + ((t & 31) * 4);
        bStepPass = 8 * ldb;    // +8 K-rows per pass (4 passes -> 32 rows)
        bStepK    = (long)KSTEP * ldb;
    }

    v8f acc[2][4];
    #pragma unroll
    for (int i = 0; i < 2; ++i)
        #pragma unroll
        for (int j = 0; j < 4; ++j) acc[i][j] = {};

    const int arow = t >> 3;        // A LDS row
    const int acol = (t & 7) * 4;   // A LDS col (float4 granule)

    for (int k0 = 0; k0 < Kdim; k0 += KSTEP) {
        // ---- stage A tile 64x32 (f32x4 loads -> packed bf16 b64 stores) ----
        #pragma unroll
        for (int p = 0; p < 4; ++p) {
            f32x4 f = *(const f32x4*)(aRun + p * lda16);
            u32x2 pk; pk.x = pk2bf(f.x, f.y); pk.y = pk2bf(f.z, f.w);
            *(u32x2*)&As[arow + 16 * p][acol] = pk;
        }
        // ---- stage B tile as Bs[n][k] ----
        if (TRANSB) {
            #pragma unroll
            for (int p = 0; p < 8; ++p) {
                f32x4 f = *(const f32x4*)(bRun + p * bStepPass);
                u32x2 pk; pk.x = pk2bf(f.x, f.y); pk.y = pk2bf(f.z, f.w);
                *(u32x2*)&Bs[(t >> 3) + 16 * p][acol] = pk;
            }
        } else {
            // load two consecutive K rows, pack vertically: one u32 per column
            const int kp = (t >> 5) * 2;    // even K row
            const int nn = (t & 31) * 4;    // 4 consecutive output cols
            #pragma unroll
            for (int p = 0; p < 4; ++p) {
                const float* b0 = bRun + p * bStepPass;
                f32x4 f0 = *(const f32x4*)(b0);
                f32x4 f1 = *(const f32x4*)(b0 + ldb);
                const int kr = kp + 8 * p;
                *(unsigned*)&Bs[nn + 0][kr] = pk2bf(f0.x, f1.x);
                *(unsigned*)&Bs[nn + 1][kr] = pk2bf(f0.y, f1.y);
                *(unsigned*)&Bs[nn + 2][kr] = pk2bf(f0.z, f1.z);
                *(unsigned*)&Bs[nn + 3][kr] = pk2bf(f0.w, f1.w);
            }
        }
        aRun += KSTEP;
        bRun += bStepK;
        if (k0 + KSTEP < Kdim) __builtin_prefetch(aRun, 0, 1);
        __syncthreads();

        // ---- fragments: 2 A + 4 B, then 8 wmma with A reuse ----
        const int aBase = 8 * rsel;    // A frag K sel per lane half
        const int bBase = 8 * rsel;    // B frag K sel per lane half
        union { u32x4 q[2]; v16bf v; } ua[2], ub[4];
        #pragma unroll
        for (int i = 0; i < 2; ++i) {
            const int ar = wr * 32 + i * 16 + rlo;
            ua[i].q[0] = *(const u32x4*)&As[ar][aBase];
            ua[i].q[1] = *(const u32x4*)&As[ar][aBase + 16];
        }
        #pragma unroll
        for (int j = 0; j < 4; ++j) {
            const int bn = wc * 64 + j * 16 + rlo;
            ub[j].q[0] = *(const u32x4*)&Bs[bn][bBase];
            ub[j].q[1] = *(const u32x4*)&Bs[bn][bBase + 16];
        }
        #pragma unroll
        for (int i = 0; i < 2; ++i)
            #pragma unroll
            for (int j = 0; j < 4; ++j)
                acc[i][j] = __builtin_amdgcn_wmma_f32_16x16x32_bf16(
                    false, ua[i].v, false, ub[j].v, (short)0, acc[i][j], false, false);
        __syncthreads();
    }

    // ---- store: f32 16x16 C/D layout: col = lane&15, row = v + 8*(lane>>4) ----
    #pragma unroll
    for (int i = 0; i < 2; ++i) {
        #pragma unroll
        for (int j = 0; j < 4; ++j) {
            const int col = n0 + wc * 64 + j * 16 + rlo;
            const float bv = bias ? bias[col] : 0.0f;
            #pragma unroll
            for (int v = 0; v < 8; ++v) {
                const int row = m0 + wr * 32 + i * 16 + v + 8 * rsel;
                float val = acc[i][j][v] * alpha + bv;
                if (RELU) val = fmaxf(val, 0.0f);
                Cb[(long)row * ldc + col] = val;
            }
        }
    }
}

// ---------------------------------------------------------------------------
// Row softmax (in place) + un-normalized combine weight wts = sum(exp(logit))
// One block (256 threads) per row; ncols = 1024.
// ---------------------------------------------------------------------------
__global__ __launch_bounds__(256)
void softmax_rows(float* __restrict__ L, float* __restrict__ wts, int ncols)
{
    __shared__ float red[256];
    const int row = blockIdx.x;
    float* Lr = L + (long)row * ncols;
    const int t = threadIdx.x;
    const int per = ncols >> 8;            // 4

    float v[4];
    float m = -3.4e38f;
    #pragma unroll
    for (int i = 0; i < 4; ++i)
        if (i < per) { v[i] = Lr[t + 256 * i]; m = fmaxf(m, v[i]); }
    red[t] = m; __syncthreads();
    for (int s = 128; s > 0; s >>= 1) { if (t < s) red[t] = fmaxf(red[t], red[t + s]); __syncthreads(); }
    const float mx = red[0]; __syncthreads();

    float s = 0.0f;
    #pragma unroll
    for (int i = 0; i < 4; ++i)
        if (i < per) { v[i] = __expf(v[i] - mx); s += v[i]; }
    red[t] = s; __syncthreads();
    for (int q = 128; q > 0; q >>= 1) { if (t < q) red[t] += red[t + q]; __syncthreads(); }
    const float sum = red[0];

    const float inv = 1.0f / sum;
    #pragma unroll
    for (int i = 0; i < 4; ++i)
        if (i < per) Lr[t + 256 * i] = v[i] * inv;
    if (t == 0) wts[row] = sum * __expf(mx);   // = sum(exp(raw logits))
}

// acc[i] = (zero ? 0 : acc[i]) + out[i] * wts[i / F]   (float4 granularity)
__global__ void accum_mix(float* __restrict__ acc, const float* __restrict__ outab,
                          const float* __restrict__ wts, int F, long total4, int zero)
{
    long i = (long)blockIdx.x * blockDim.x + threadIdx.x;
    if (i < total4) {
        long e   = i * 4;
        long tok = e / F;                 // F multiple of 4: one token per float4
        float w  = wts[tok];
        f32x4 o = *(const f32x4*)(outab + e);
        f32x4 a;
        if (zero) a = (f32x4){0.f, 0.f, 0.f, 0.f};
        else      a = *(const f32x4*)(acc + e);
        a.x += o.x * w; a.y += o.y * w; a.z += o.z * w; a.w += o.w * w;
        *(f32x4*)(acc + e) = a;
    }
}

// sumw[t] = sum_b |wts_b[t]|
__global__ void sum_wts(const float* __restrict__ wts, float* __restrict__ sumw, int T)
{
    int i = blockIdx.x * blockDim.x + threadIdx.x;
    if (i < T)
        sumw[i] = fabsf(wts[i]) + fabsf(wts[T + i]) + fabsf(wts[2 * T + i]) + fabsf(wts[3 * T + i]);
}

// ---------------------------------------------------------------------------
// Fused (base + add*inv) -> LayerNorm -> out.   inv = 1/max(sumw,1e-12) when
// sumw != null (mix path), else 1 (residual path). One block per row.
// ---------------------------------------------------------------------------
__global__ __launch_bounds__(256)
void add_layernorm(const float* __restrict__ base, const float* __restrict__ add,
                   const float* __restrict__ sumw,
                   const float* __restrict__ g, const float* __restrict__ bb,
                   float* __restrict__ out, int F)
{
    __shared__ float red[256];
    const int row = blockIdx.x;
    const int t = threadIdx.x;
    const int per = F >> 8;                // 1 or 2
    const long rb = (long)row * F;

    const float inv = sumw ? (1.0f / fmaxf(sumw[row], 1e-12f)) : 1.0f;

    float x[2];
    float s = 0.0f;
    #pragma unroll
    for (int i = 0; i < 2; ++i)
        if (i < per) {
            int c = t + 256 * i;
            x[i] = base[rb + c] + add[rb + c] * inv;
            s += x[i];
        }
    red[t] = s; __syncthreads();
    for (int q = 128; q > 0; q >>= 1) { if (t < q) red[t] += red[t + q]; __syncthreads(); }
    const float mu = red[0] / (float)F; __syncthreads();

    float s2 = 0.0f;
    #pragma unroll
    for (int i = 0; i < 2; ++i)
        if (i < per) { float d = x[i] - mu; s2 += d * d; }
    red[t] = s2; __syncthreads();
    for (int q = 128; q > 0; q >>= 1) { if (t < q) red[t] += red[t + q]; __syncthreads(); }
    const float var = red[0] / (float)F;
    const float rs = rsqrtf(var + 1e-7f);

    #pragma unroll
    for (int i = 0; i < 2; ++i)
        if (i < per) {
            int c = t + 256 * i;
            out[rb + c] = (x[i] - mu) * rs * g[c] + bb[c];
        }
}

// ---------------------------------------------------------------------------
// Host orchestration
// ---------------------------------------------------------------------------
extern "C" void kernel_launch(void* const* d_in, const int* in_sizes, int n_in,
                              void* d_out, int out_size, void* d_ws, size_t ws_size,
                              hipStream_t stream)
{
    (void)in_sizes; (void)n_in; (void)out_size; (void)ws_size;

    const int GFv[4] = {512, 256, 512, 256};
    const int FFv[4] = {1024, 512, 1024, 512};
    const int Bb = 8, Nn = 1024;
    const int T = Bb * Nn;                       // 8192 tokens

    // ---- input pytree flattening ----
    // top-level dict order: w0,w1,w2,w3, params
    // params (jax sorted): attn[(a,b) sorted] each {Wk,Wo,Wq,Wv,bk,bo,bq,bv},
    // ffn[a] {W1,W2,b1,b2}, ln1[a] {b,g}, ln2[a] {b,g}
    auto IN = [&](int i) { return (const float*)d_in[i]; };
    const int attnBase = 4;
    auto aidx = [&](int a, int b, int leaf) { return attnBase + (a * 4 + b) * 8 + leaf; };
    const int L_Wk = 0, L_Wo = 1, L_Wq = 2, L_Wv = 3, L_bk = 4, L_bo = 5, L_bq = 6, L_bv = 7;
    const int ffnBase = attnBase + 16 * 8;       // 132
    const int ln1Base = ffnBase + 16;            // 148
    const int ln2Base = ln1Base + 8;             // 156

    // ---- workspace carve-up (floats) ----
    float* ws = (float*)d_ws;
    size_t off = 0;
    float* logits = ws + off; off += (size_t)Bb * Nn * Nn;   // 8M (reused as FFN hidden)
    float* Qb     = ws + off; off += (size_t)T * 256;
    float* Kb     = ws + off; off += (size_t)T * 256;
    float* Vb     = ws + off; off += (size_t)T * 256;
    float* Cx     = ws + off; off += (size_t)T * 256;        // attn @ V
    float* outab  = ws + off; off += (size_t)T * 512;        // reused as FFN y
    float* accb   = ws + off; off += (size_t)T * 512;
    float* xb     = ws + off; off += (size_t)T * 512;
    float* wtsb   = ws + off; off += (size_t)4 * T;
    float* sumwb  = ws + off; off += (size_t)T;
    float* hb = logits;      // FFN hidden reuses logits buffer
    float* yb = outab;       // FFN output reuses outab buffer

    size_t outOff[4];
    outOff[0] = 0;
    for (int a = 1; a < 4; ++a) outOff[a] = outOff[a - 1] + (size_t)T * GFv[a - 1];
    float* Out = (float*)d_out;

    const dim3 blk128(128), blk256(256);

    for (int a = 0; a < 4; ++a) {
        const int Fa = GFv[a];
        for (int b = 0; b < 4; ++b) {
            const int Fb = GFv[b];
            const int di = (Fa < Fb ? Fa : Fb) / 2;   // 128 or 256
            const float* Wq = IN(aidx(a, b, L_Wq)); const float* bq = IN(aidx(a, b, L_bq));
            const float* Wk = IN(aidx(a, b, L_Wk)); const float* bk = IN(aidx(a, b, L_bk));
            const float* Wv = IN(aidx(a, b, L_Wv)); const float* bv = IN(aidx(a, b, L_bv));
            const float* Wo = IN(aidx(a, b, L_Wo)); const float* bo = IN(aidx(a, b, L_bo));

            // Q = w_a @ Wq + bq       (T x di)
            gemm_wmma_bf16<false, false><<<dim3(di / BN, T / BM, 1), blk128, 0, stream>>>(
                IN(a), Fa, 0, Wq, di, 0, bq, Qb, di, 0, T, di, Fa, 1.0f);
            // K = w_b @ Wk + bk
            gemm_wmma_bf16<false, false><<<dim3(di / BN, T / BM, 1), blk128, 0, stream>>>(
                IN(b), Fb, 0, Wk, di, 0, bk, Kb, di, 0, T, di, Fb, 1.0f);
            // V = w_b @ Wv + bv
            gemm_wmma_bf16<false, false><<<dim3(di / BN, T / BM, 1), blk128, 0, stream>>>(
                IN(b), Fb, 0, Wv, di, 0, bv, Vb, di, 0, T, di, Fb, 1.0f);

            // logits = Q @ K^T / sqrt(di)   batched over Bb
            gemm_wmma_bf16<true, false><<<dim3(Nn / BN, Nn / BM, Bb), blk128, 0, stream>>>(
                Qb, di, (long)Nn * di, Kb, di, (long)Nn * di, nullptr,
                logits, Nn, (long)Nn * Nn, Nn, Nn, di, 1.0f / sqrtf((float)di));

            // softmax rows + combine weight
            softmax_rows<<<dim3(T), blk256, 0, stream>>>(logits, wtsb + (size_t)b * T, Nn);

            // ctx = attn @ V   batched
            gemm_wmma_bf16<false, false><<<dim3(di / BN, Nn / BM, Bb), blk128, 0, stream>>>(
                logits, Nn, (long)Nn * Nn, Vb, di, (long)Nn * di, nullptr,
                Cx, di, (long)Nn * di, Nn, di, Nn, 1.0f);

            // out_ab = ctx @ Wo + bo     (T x Fa)
            gemm_wmma_bf16<false, false><<<dim3(Fa / BN, T / BM, 1), blk128, 0, stream>>>(
                Cx, di, 0, Wo, Fa, 0, bo, outab, Fa, 0, T, Fa, di, 1.0f);

            // acc += out_ab * wts_b
            long total4 = (long)T * Fa / 4;
            accum_mix<<<dim3((unsigned)((total4 + 255) / 256)), blk256, 0, stream>>>(
                accb, outab, wtsb + (size_t)b * T, Fa, total4, b == 0 ? 1 : 0);
        }

        // denom and mix + residual + LN1 -> x
        sum_wts<<<dim3(T / 256), blk256, 0, stream>>>(wtsb, sumwb, T);
        add_layernorm<<<dim3(T), blk256, 0, stream>>>(
            IN(a), accb, sumwb, IN(ln1Base + a * 2 + 1), IN(ln1Base + a * 2 + 0), xb, Fa);

        // FFN: h = relu(x@W1 + b1); y = h@W2 + b2
        const float* W1 = IN(ffnBase + a * 4 + 0); const float* W2 = IN(ffnBase + a * 4 + 1);
        const float* b1 = IN(ffnBase + a * 4 + 2); const float* b2 = IN(ffnBase + a * 4 + 3);
        const int Ha = FFv[a];
        gemm_wmma_bf16<false, true><<<dim3(Ha / BN, T / BM, 1), blk128, 0, stream>>>(
            xb, Fa, 0, W1, Ha, 0, b1, hb, Ha, 0, T, Ha, Fa, 1.0f);
        gemm_wmma_bf16<false, false><<<dim3(Fa / BN, T / BM, 1), blk128, 0, stream>>>(
            hb, Ha, 0, W2, Fa, 0, b2, yb, Fa, 0, T, Fa, Ha, 1.0f);

        // out_a = LN2(x + y)
        add_layernorm<<<dim3(T), blk256, 0, stream>>>(
            xb, yb, nullptr, IN(ln2Base + a * 2 + 1), IN(ln2Base + a * 2 + 0),
            Out + outOff[a], Fa);
    }
}
